// TransformerGraphEventDecoderBlock_50044958933427
// MI455X (gfx1250) — matrix operand order repaired and checked by reading, not compiled, not run on hardware
//
// MI455X / gfx1250 implementation of TransformerGraphEventDecoderBlock.
//
// Roofline: ~583 GFLOP of K/V-projection GEMMs dominate; ~650 MB of HBM
// traffic (~28 us @ 23.3 TB/s) -> compute bound at bf16 WMMA rates, so all
// GEMMs run through V_WMMA_F32_16X16X32_BF16 with fp32 accumulation.
//
// CDNA5-specific paths used:
//  - v_wmma_f32_16x16x32_bf16 (64 per K-step per block, 2x4 accum per wave)
//  - global_load_async_to_lds_b128 + s_wait_asynccnt for tile staging
//    (operands pre-converted to bf16; weights pre-transposed to [N,K] so both
//     A and B tiles are contiguous 64B rows landing directly in the WMMA
//     fragment layout)
//  - ds_load pairs for fragment gathers per ISA 7.12.2 wave32 layouts
//
// Deterministic (no atomics), all launches on `stream`, no allocation.
// Workspace ~330 MB.

#include <hip/hip_runtime.h>
#include <hip/hip_bf16.h>
#include <stdint.h>

typedef __attribute__((ext_vector_type(16))) __bf16 v16bf;
typedef __attribute__((ext_vector_type(8)))  float  v8f;

// ---------------------------------------------------------------- utilities
__device__ __forceinline__ unsigned short f2bf(float f) {
  unsigned int u = __float_as_uint(f);
  u += 0x7FFFu + ((u >> 16) & 1u);          // round-to-nearest-even
  return (unsigned short)(u >> 16);
}
__device__ __forceinline__ unsigned int pack2(float lo, float hi) {
  return (unsigned int)f2bf(lo) | ((unsigned int)f2bf(hi) << 16);
}
__device__ __forceinline__ float bf2f(unsigned short h) {
  return __uint_as_float(((unsigned int)h) << 16);
}

// =================================================================== GEMM
// C[M,N] = A[M,K](bf16, row stride lda) * Wt[N,K](bf16, pre-transposed)
// epilogue: +bias; valid-row zeroing; +residual(fp32); relu; store fp32 C
// and/or bf16 Cbf.
//
// Tiles staged with GLOBAL_LOAD_ASYNC_TO_LDS_B128 (ASYNCcnt). LDS rows are
// 64B of data + 16B pad (stride 80B) so fragment reads are bank-spread while
// async chunks stay 16B-aligned.
__global__ __launch_bounds__(256)
void gemm_bf16_async(const unsigned short* __restrict__ A, long lda,
                     const unsigned short* __restrict__ Wt,
                     float* __restrict__ C, long ldc,
                     unsigned short* __restrict__ Cbf, long ldcbf,
                     int M, int N, int K,
                     const float* __restrict__ bias,
                     const float* __restrict__ residual, long ldres,
                     const int* __restrict__ validRow, int applyRelu) {
  __shared__ unsigned char smem[2 * 128 * 80];      // As | Bs
  const unsigned int* As32 = (const unsigned int*)smem;
  const unsigned int* Bs32 = (const unsigned int*)(smem + 128 * 80);
  const unsigned int ldsA = (unsigned int)(uintptr_t)(void*)smem;
  const unsigned int ldsB = ldsA + 128 * 80;

  const int tid  = threadIdx.x;
  const int lane = tid & 31;
  const int wave = tid >> 5;        // 8 waves
  const int wm   = wave & 3;        // 4 x 32 rows
  const int wn   = wave >> 2;       // 2 x 64 cols
  const long blockM = (long)blockIdx.x * 128;
  const long blockN = (long)blockIdx.y * 128;

  // per-thread async-copy slots: 2x 16B chunks for A, 2x for B per K-step.
  // chunk c = tid + i*256 in [0,512): row = c>>2, seg = c&3 (4 x 16B = 64B row)
  unsigned long long aAddr[2], bAddr[2];
  unsigned int aLds[2], bLds[2];
  bool aOk[2];
#pragma unroll
  for (int i = 0; i < 2; ++i) {
    int c = tid + i * 256;
    int row = c >> 2, seg = c & 3;
    aOk[i]   = (blockM + row) < M;
    aAddr[i] = (unsigned long long)(uintptr_t)(A + (blockM + row) * lda) + seg * 16;
    aLds[i]  = ldsA + row * 80 + seg * 16;
    bAddr[i] = (unsigned long long)(uintptr_t)(Wt + (blockN + row) * (long)K) + seg * 16;
    bLds[i]  = ldsB + row * 80 + seg * 16;
  }

  v8f acc[2][4];
#pragma unroll
  for (int mt = 0; mt < 2; ++mt)
#pragma unroll
    for (int nt = 0; nt < 4; ++nt)
#pragma unroll
      for (int r = 0; r < 8; ++r) acc[mt][nt][r] = 0.0f;

  union Frag { unsigned int u[8]; v16bf v; };

  for (int k0 = 0; k0 < K; k0 += 32) {
    // ---- async-stage A (128x32 bf16) and B (128 cols x 32 k) tiles
#pragma unroll
    for (int i = 0; i < 2; ++i) {
      if (aOk[i])
        asm volatile("global_load_async_to_lds_b128 %0, %1, off"
                     :: "v"(aLds[i]), "v"(aAddr[i]) : "memory");
      asm volatile("global_load_async_to_lds_b128 %0, %1, off"
                   :: "v"(bLds[i]), "v"(bAddr[i]) : "memory");
    }
    asm volatile("s_wait_asynccnt 0x0" ::: "memory");
    __syncthreads();

    // ---- fragments per ISA 7.12.2 (wave32)
    Frag afr[2], bfr[4];
    const int klo4 = (lane >> 4) * 4;       // A: lanes16-31 start at K=8
#pragma unroll
    for (int mt = 0; mt < 2; ++mt) {
      int m = wm * 32 + mt * 16 + (lane & 15);
#pragma unroll
      for (int v = 0; v < 8; ++v)
        afr[mt].u[v] = As32[m * 20 + klo4 + (v & 3) + ((v >> 2) << 3)];
    }
    const int kb = (lane >> 4) * 4;         // B: lanes16-31 start at K=8
#pragma unroll
    for (int nt = 0; nt < 4; ++nt) {
      int n = wn * 64 + nt * 16 + (lane & 15);
#pragma unroll
      for (int e = 0; e < 8; ++e) bfr[nt].u[e] = Bs32[n * 20 + kb + (e & 3) + ((e >> 2) << 3)];
    }

#pragma unroll
    for (int mt = 0; mt < 2; ++mt)
#pragma unroll
      for (int nt = 0; nt < 4; ++nt)
        acc[mt][nt] = __builtin_amdgcn_wmma_f32_16x16x32_bf16(
            false, afr[mt].v, false, bfr[nt].v, (short)0, acc[mt][nt],
            false, false);
    __syncthreads();          // fragments consumed before next overwrite

#pragma unroll
    for (int i = 0; i < 2; ++i) { aAddr[i] += 64; bAddr[i] += 64; }
  }

  // ---- epilogue (C/D layout: VGPR r -> M=(lane/16)*8+r, N=lane%16)
#pragma unroll
  for (int mt = 0; mt < 2; ++mt)
#pragma unroll
    for (int nt = 0; nt < 4; ++nt) {
      long rowBase = blockM + wm * 32 + mt * 16 + ((lane >> 4) << 3);
      long col     = blockN + wn * 64 + nt * 16 + (lane & 15);
#pragma unroll
      for (int r = 0; r < 8; ++r) {
        long row = rowBase + r;
        if (row < M) {
          float v = acc[mt][nt][r];
          if (bias) v += bias[col];
          if (validRow && !validRow[row]) v = 0.0f;
          if (residual) v += residual[row * ldres + col];
          if (applyRelu) v = fmaxf(v, 0.0f);
          if (C)   C[row * ldc + col] = v;
          if (Cbf) Cbf[row * ldcbf + col] = f2bf(v);
        }
      }
    }
}

// ============================================== conversion / transposition
// elementwise fp32 -> bf16 (n4 = element count / 4)
__global__ __launch_bounds__(256)
void f32_to_bf16_kernel(const float* __restrict__ x,
                        unsigned short* __restrict__ y, long n4) {
  long i = (long)blockIdx.x * 256 + threadIdx.x;
  long stride = (long)gridDim.x * 256;
  unsigned int* yo = (unsigned int*)y;
  for (; i < n4; i += stride) {
    float4 v = ((const float4*)x)[i];
    yo[i * 2 + 0] = pack2(v.x, v.y);
    yo[i * 2 + 1] = pack2(v.z, v.w);
  }
}

// W[K,N] fp32 -> Wt[N,K] bf16 (32x32 LDS tiles; K,N multiples of 32)
__global__ __launch_bounds__(256)
void transpose_f32_to_bf16(const float* __restrict__ W, int K, int N,
                           unsigned short* __restrict__ Wt) {
  __shared__ float tile[32][33];
  int n0 = blockIdx.x * 32, k0 = blockIdx.y * 32;
  for (int kk = threadIdx.y; kk < 32; kk += 8)
    tile[kk][threadIdx.x] = W[(long)(k0 + kk) * N + n0 + threadIdx.x];
  __syncthreads();
  for (int nn = threadIdx.y; nn < 32; nn += 8)
    Wt[(long)(n0 + nn) * K + k0 + threadIdx.x] = f2bf(tile[threadIdx.x][nn]);
}

// ============================================================= LayerNorm
__device__ __forceinline__ void ln_row_body(const float* __restrict__ src,
                                            const float* __restrict__ g,
                                            const float* __restrict__ b,
                                            unsigned short* __restrict__ dst) {
  __shared__ float red[256];
  const int t = threadIdx.x;
  float v[4];
  float s = 0.f;
#pragma unroll
  for (int i = 0; i < 4; ++i) { v[i] = src[t + i * 256]; s += v[i]; }
  red[t] = s; __syncthreads();
  for (int st = 128; st > 0; st >>= 1) { if (t < st) red[t] += red[t + st]; __syncthreads(); }
  float mean = red[0] * (1.f / 1024.f); __syncthreads();
  float vs = 0.f;
#pragma unroll
  for (int i = 0; i < 4; ++i) { float d = v[i] - mean; vs += d * d; }
  red[t] = vs; __syncthreads();
  for (int st = 128; st > 0; st >>= 1) { if (t < st) red[t] += red[t + st]; __syncthreads(); }
  float rstd = rsqrtf(red[0] * (1.f / 1024.f) + 1e-5f); __syncthreads();
#pragma unroll
  for (int i = 0; i < 4; ++i) {
    int idx = t + i * 256;
    dst[idx] = f2bf((v[i] - mean) * rstd * g[idx] + b[idx]);
  }
}

// concat(prev_seq[B,511,H], input[B,1,H]) -> LN -> normed bf16 [B,512,H]
__global__ __launch_bounds__(256)
void ln1_kernel(const float* __restrict__ prev, const float* __restrict__ input,
                const float* __restrict__ g, const float* __restrict__ b,
                unsigned short* __restrict__ out) {
  int row = blockIdx.x;                 // b*512 + l
  int bb = row >> 9, l = row & 511;
  const float* src = (l < 511) ? (prev + ((long)bb * 511 + l) * 1024)
                               : (input + (long)bb * 1024);
  ln_row_body(src, g, b, out + (long)row * 1024);
}

__global__ __launch_bounds__(256)
void ln_rows_kernel(const float* __restrict__ x, const float* __restrict__ g,
                    const float* __restrict__ b,
                    unsigned short* __restrict__ y) {
  ln_row_body(x + (long)blockIdx.x * 1024, g, b, y + (long)blockIdx.x * 1024);
}

// ===================================================== attention epilogues
// scores + softmax per (b, head); 1 query; K stored bf16; L <= 512
__global__ __launch_bounds__(256)
void scores_softmax_kernel(const float* __restrict__ q,
                           const unsigned short* __restrict__ Kmat,
                           const uint8_t* __restrict__ mask, int L,
                           float* __restrict__ wout) {
  const int b = blockIdx.x, h = blockIdx.y, t = threadIdx.x;
  __shared__ float qs[64];
  __shared__ float red[256];
  if (t < 64) qs[t] = q[b * 1024 + h * 64 + t];
  __syncthreads();
  float sv[2]; sv[0] = sv[1] = -3.4e38f;
  float mymax = -3.4e38f;
#pragma unroll
  for (int j = 0; j < 2; ++j) {
    int l = t + j * 256;
    if (l < L) {
      const unsigned short* kr = Kmat + ((long)b * L + l) * 1024 + h * 64;
      float d = 0.f;
#pragma unroll
      for (int i = 0; i < 64; ++i) d += qs[i] * bf2f(kr[i]);
      d *= 0.125f;                                 // 1/sqrt(64)
      if (!mask[(long)b * L + l]) d = -1.0e9f;     // NEG
      sv[j] = d;
      mymax = fmaxf(mymax, d);
    }
  }
  red[t] = mymax; __syncthreads();
  for (int st = 128; st > 0; st >>= 1) { if (t < st) red[t] = fmaxf(red[t], red[t + st]); __syncthreads(); }
  float mx = red[0]; __syncthreads();
  float ev[2] = {0.f, 0.f};
  float se = 0.f;
#pragma unroll
  for (int j = 0; j < 2; ++j) {
    int l = t + j * 256;
    if (l < L) { ev[j] = __expf(sv[j] - mx); se += ev[j]; }
  }
  red[t] = se; __syncthreads();
  for (int st = 128; st > 0; st >>= 1) { if (t < st) red[t] += red[t + st]; __syncthreads(); }
  float inv = 1.f / red[0];
#pragma unroll
  for (int j = 0; j < 2; ++j) {
    int l = t + j * 256;
    if (l < L) wout[((long)b * 16 + h) * 512 + l] = ev[j] * inv;
  }
}

// out[b, h*64+d] = sum_l w[b,h,l] * V[b,l,h,d]  (V bf16, out bf16)
__global__ __launch_bounds__(64)
void weighted_v_kernel(const float* __restrict__ w,
                       const unsigned short* __restrict__ V, int L,
                       unsigned short* __restrict__ out) {
  const int b = blockIdx.x, h = blockIdx.y, d = threadIdx.x;
  const float* wr = w + ((long)b * 16 + h) * 512;
  const unsigned short* vp = V + (long)b * L * 1024 + h * 64 + d;
  float acc = 0.f;
  for (int l = 0; l < L; ++l) acc += wr[l] * bf2f(vp[(long)l * 1024]);
  out[b * 1024 + h * 64 + d] = f2bf(acc);
}

// deterministic mean over heads (optionally masked by per-batch valid flag)
__global__ __launch_bounds__(256)
void mean_w_kernel(const float* __restrict__ w, int L,
                   const int* __restrict__ valid, float* __restrict__ out) {
  const int b = blockIdx.x;
  float scale = (1.f / 16.f);
  if (valid && !valid[b]) scale = 0.f;
  for (int l = threadIdx.x; l < L; l += 256) {
    float s = 0.f;
#pragma unroll
    for (int h = 0; h < 16; ++h) s += w[((long)b * 16 + h) * 512 + l];
    out[(long)b * L + l] = s * scale;
  }
}

__global__ __launch_bounds__(256)
void valid_kernel(const uint8_t* __restrict__ nm, int* __restrict__ valid) {
  __shared__ int red[256];
  const int b = blockIdx.x, t = threadIdx.x;
  int any = 0;
  for (int l = t; l < 512; l += 256) any |= (nm[b * 512 + l] != 0);
  red[t] = any; __syncthreads();
  for (int st = 128; st > 0; st >>= 1) { if (t < st) red[t] |= red[t + st]; __syncthreads(); }
  if (t == 0) valid[b] = red[0];
}

__global__ __launch_bounds__(256)
void build_cmask_kernel(const uint8_t* __restrict__ prevm,
                        const uint8_t* __restrict__ inm,
                        uint8_t* __restrict__ cm) {
  const int b = blockIdx.x;
  for (int l = threadIdx.x; l < 512; l += 256)
    cm[b * 512 + l] = (l < 511) ? prevm[b * 511 + l] : inm[b];
}

// =================================================================== host
static inline const float* PF(void* const* d, int i) { return (const float*)d[i]; }

static void launch_gemm(const unsigned short* A, long lda,
                        const unsigned short* Wt, float* C, long ldc,
                        unsigned short* Cbf, long ldcbf, int M, int N, int K,
                        const float* bias, const float* resid, long ldres,
                        const int* valid, int relu, hipStream_t s) {
  dim3 grid((M + 127) / 128, N / 128);
  gemm_bf16_async<<<grid, 256, 0, s>>>(A, lda, Wt, C, ldc, Cbf, ldcbf, M, N, K,
                                       bias, resid, ldres, valid, relu);
}

// one full MHA (1 query token): q/K/V proj, softmax, weighted-V, out proj
static void run_attn(void* const* d_in, int wbase,
                     const unsigned short* qrowsBF, long ldq,
                     const unsigned short* kvBF, int L, const uint8_t* mask,
                     const unsigned short* WtBase,
                     float* Cout, long ldc, unsigned short* CbfOut, long ldcbf,
                     const float* resid, const int* valid, float* meanDst,
                     float* QBUF, unsigned short* KBF, unsigned short* VBF,
                     float* WBUF, unsigned short* HCATBF, hipStream_t s) {
  const float* bq = PF(d_in, wbase + 1);
  const float* bk = PF(d_in, wbase + 3);
  const float* bv = PF(d_in, wbase + 5);
  const float* bo = PF(d_in, wbase + 7);
  const unsigned short* WtQ = WtBase;
  const unsigned short* WtK = WtBase + 1 * 1048576;
  const unsigned short* WtV = WtBase + 2 * 1048576;
  const unsigned short* WtO = WtBase + 3 * 1048576;
  launch_gemm(qrowsBF, ldq, WtQ, QBUF, 1024, nullptr, 0, 64, 1024, 1024, bq,
              nullptr, 0, nullptr, 0, s);
  launch_gemm(kvBF, 1024, WtK, nullptr, 0, KBF, 1024, 64 * L, 1024, 1024, bk,
              nullptr, 0, nullptr, 0, s);
  launch_gemm(kvBF, 1024, WtV, nullptr, 0, VBF, 1024, 64 * L, 1024, 1024, bv,
              nullptr, 0, nullptr, 0, s);
  scores_softmax_kernel<<<dim3(64, 16), 256, 0, s>>>(QBUF, KBF, mask, L, WBUF);
  weighted_v_kernel<<<dim3(64, 16), 64, 0, s>>>(WBUF, VBF, L, HCATBF);
  launch_gemm(HCATBF, 1024, WtO, Cout, ldc, CbfOut, ldcbf, 64, 1024, 1024, bo,
              resid, 1024, valid, 0, s);
  mean_w_kernel<<<64, 256, 0, s>>>(WBUF, L, valid, meanDst);
}

extern "C" void kernel_launch(void* const* d_in, const int* in_sizes, int n_in,
                              void* d_out, int out_size, void* d_ws,
                              size_t ws_size, hipStream_t stream) {
  (void)in_sizes; (void)n_in; (void)out_size; (void)ws_size;

  // ---- inputs (setup_inputs order; params flattened in dict insertion order)
  const float*   input      = (const float*)d_in[0];
  const uint8_t* input_mask = (const uint8_t*)d_in[1];
  const float*   obs_kv     = (const float*)d_in[2];
  const uint8_t* obs_mask   = (const uint8_t*)d_in[3];
  const float*   act_kv     = (const float*)d_in[4];
  const uint8_t* act_mask   = (const uint8_t*)d_in[5];
  const float*   gobs_kv    = (const float*)d_in[6];
  const float*   gact_kv    = (const float*)d_in[7];
  const uint8_t* node_mask  = (const uint8_t*)d_in[8];
  const float*   prev_seq   = (const float*)d_in[9];
  const uint8_t* prev_mask  = (const uint8_t*)d_in[10];
  // params: 11 ln1_g, 12 ln1_b, 13..20 sa{Wq,bq,Wk,bk,Wv,bv,Wo,bo},
  // 21 ln2_g, 22 ln2_b, 23..30 obs, 31..38 act, 39..46 gobs, 47..54 gact,
  // 55 Wc, 56 bc, 57 ln3_g, 58 ln3_b, 59 W1, 60 b1, 61 W2, 62 b2

  float* out = (float*)d_out;
  float* out_main = out;              // (64,1024)
  float* out_saw  = out + 65536;      // (64,512)
  float* out_obsw = out + 98304;      // (64,512)
  float* out_actw = out + 131072;     // (64,128)
  float* out_gow  = out + 139264;     // (64,512)
  float* out_gaw  = out + 172032;     // (64,512)

  // ---- workspace carve-up (float units)
  float* ws = (float*)d_ws;
  size_t o = 0;
  unsigned short* NORMBF = (unsigned short*)(ws + o); o += (size_t)64 * 512 * 1024 / 2;
  unsigned short* KVBF   = (unsigned short*)(ws + o); o += (size_t)64 * 512 * 1024 / 2;
  unsigned short* KBF    = (unsigned short*)(ws + o); o += (size_t)64 * 512 * 1024 / 2;
  unsigned short* VBF    = (unsigned short*)(ws + o); o += (size_t)64 * 512 * 1024 / 2;
  unsigned short* WT     = (unsigned short*)(ws + o); o += (size_t)27262976 / 2;
  float* QBUF    = ws + o; o += 64 * 1024;
  float* WBUF    = ws + o; o += (size_t)64 * 16 * 512;
  float* SABUF   = ws + o; o += 64 * 1024;
  float* COMB    = ws + o; o += 64 * 1024;
  unsigned short* QUERYBF = (unsigned short*)(ws + o); o += 64 * 1024 / 2;
  unsigned short* HCATBF  = (unsigned short*)(ws + o); o += 64 * 1024 / 2;
  unsigned short* CAT4BF  = (unsigned short*)(ws + o); o += (size_t)64 * 4096 / 2;
  unsigned short* LN3BF   = (unsigned short*)(ws + o); o += 64 * 1024 / 2;
  unsigned short* H1BF    = (unsigned short*)(ws + o); o += 64 * 1024 / 2;
  int*     VALID = (int*)(ws + o); o += 64;
  uint8_t* CMASK = (uint8_t*)(ws + o); o += 8192 / 4;

  // transposed bf16 weights: 5 attn blocks of 4x(1024x1024), Wc, W1, W2
  unsigned short* WtSA   = WT;
  unsigned short* WtOBS  = WT + (size_t)4 * 1048576;
  unsigned short* WtACT  = WT + (size_t)8 * 1048576;
  unsigned short* WtGOBS = WT + (size_t)12 * 1048576;
  unsigned short* WtGACT = WT + (size_t)16 * 1048576;
  unsigned short* WtC    = WT + (size_t)20 * 1048576;   // 1024 x 4096
  unsigned short* WtW1   = WtC + (size_t)4 * 1048576;
  unsigned short* WtW2   = WtW1 + (size_t)1048576;

  // ---- weight convert+transpose passes (W[K,N] fp32 -> Wt[N,K] bf16)
  const int abase[5] = {13, 23, 31, 39, 47};
  unsigned short* wtbase[5] = {WtSA, WtOBS, WtACT, WtGOBS, WtGACT};
  for (int a = 0; a < 5; ++a)
    for (int widx = 0; widx < 4; ++widx)   // Wq, Wk, Wv, Wo at +0,+2,+4,+6
      transpose_f32_to_bf16<<<dim3(32, 32), dim3(32, 8), 0, stream>>>(
          PF(d_in, abase[a] + 2 * widx), 1024, 1024,
          wtbase[a] + (size_t)widx * 1048576);
  transpose_f32_to_bf16<<<dim3(32, 128), dim3(32, 8), 0, stream>>>(
      PF(d_in, 55), 4096, 1024, WtC);
  transpose_f32_to_bf16<<<dim3(32, 32), dim3(32, 8), 0, stream>>>(
      PF(d_in, 59), 1024, 1024, WtW1);
  transpose_f32_to_bf16<<<dim3(32, 32), dim3(32, 8), 0, stream>>>(
      PF(d_in, 61), 1024, 1024, WtW2);

  // ---- prologue
  valid_kernel<<<64, 256, 0, stream>>>(node_mask, VALID);
  build_cmask_kernel<<<64, 256, 0, stream>>>(prev_mask, input_mask, CMASK);
  ln1_kernel<<<64 * 512, 256, 0, stream>>>(prev_seq, input, PF(d_in, 11),
                                           PF(d_in, 12), NORMBF);

  // ---- self attention (query = last normed row; residual = input row)
  run_attn(d_in, 13, NORMBF + (long)511 * 1024, (long)512 * 1024, NORMBF, 512,
           CMASK, WtSA, SABUF, 1024, nullptr, 0, input, nullptr, out_saw,
           QBUF, KBF, VBF, WBUF, HCATBF, stream);
  ln_rows_kernel<<<64, 256, 0, stream>>>(SABUF, PF(d_in, 21), PF(d_in, 22),
                                         QUERYBF);

  // ---- cross attentions (residual = sa; gobs/gact zeroed when !valid)
  f32_to_bf16_kernel<<<8192, 256, 0, stream>>>(obs_kv, KVBF, (long)64 * 512 * 256);
  run_attn(d_in, 23, QUERYBF, 1024, KVBF, 512, obs_mask, WtOBS, nullptr, 0,
           CAT4BF + 0, 4096, SABUF, nullptr, out_obsw, QBUF, KBF, VBF, WBUF,
           HCATBF, stream);
  f32_to_bf16_kernel<<<8192, 256, 0, stream>>>(act_kv, KVBF, (long)64 * 128 * 256);
  run_attn(d_in, 31, QUERYBF, 1024, KVBF, 128, act_mask, WtACT, nullptr, 0,
           CAT4BF + 1024, 4096, SABUF, nullptr, out_actw, QBUF, KBF, VBF, WBUF,
           HCATBF, stream);
  f32_to_bf16_kernel<<<8192, 256, 0, stream>>>(gobs_kv, KVBF, (long)64 * 512 * 256);
  run_attn(d_in, 39, QUERYBF, 1024, KVBF, 512, node_mask, WtGOBS, nullptr, 0,
           CAT4BF + 2048, 4096, SABUF, VALID, out_gow, QBUF, KBF, VBF, WBUF,
           HCATBF, stream);
  f32_to_bf16_kernel<<<8192, 256, 0, stream>>>(gact_kv, KVBF, (long)64 * 512 * 256);
  run_attn(d_in, 47, QUERYBF, 1024, KVBF, 512, node_mask, WtGACT, nullptr, 0,
           CAT4BF + 3072, 4096, SABUF, VALID, out_gaw, QBUF, KBF, VBF, WBUF,
           HCATBF, stream);

  // ---- combine + MLP
  launch_gemm(CAT4BF, 4096, WtC, COMB, 1024, nullptr, 0, 64, 1024, 4096,
              PF(d_in, 56), nullptr, 0, nullptr, 1, stream);        // relu
  ln_rows_kernel<<<64, 256, 0, stream>>>(COMB, PF(d_in, 57), PF(d_in, 58),
                                         LN3BF);
  launch_gemm(LN3BF, 1024, WtW1, nullptr, 0, H1BF, 1024, 64, 1024, 1024,
              PF(d_in, 60), nullptr, 0, nullptr, 1, stream);        // relu
  launch_gemm(H1BF, 1024, WtW2, out_main, 1024, nullptr, 0, 64, 1024, 1024,
              PF(d_in, 62), COMB, 1024, nullptr, 0, stream);        // +combined
}
